// AttentionBlock_7078106103951
// MI455X (gfx1250) — compile-verified, matrix operand order checked by reading, not run on hardware
//
#include <hip/hip_runtime.h>
#include <hip/hip_bf16.h>

// Attention block for MI455X (gfx1250), wave32 + WMMA f16 (f32 accum) +
// async global->LDS (ASYNCcnt) staging of K/V tiles in the flash kernel.
// Workspace use: 256 B stats + q/k/v (3 x 8 MB f16) + attn_out (8 MB f16) ~ 33 MB.

typedef _Float16 v16h __attribute__((ext_vector_type(16)));
typedef _Float16 v8h  __attribute__((ext_vector_type(8)));
typedef float    v8f  __attribute__((ext_vector_type(8)));

#define BB     4
#define CC     256
#define NN     4096
#define NHEADS 4
#define HD     64
#define NGROUP 8

static __device__ __forceinline__ v8f wmma16(v16h a, v16h b, v8f c) {
    return __builtin_amdgcn_wmma_f32_16x16x32_f16(
        /*neg_a=*/false, a, /*neg_b=*/false, b,
        /*c_mod=*/(short)0, c, /*reuse_a=*/false, /*reuse_b=*/false);
}

// 16x32 f16 fragment load (A operand: lane=row; B operand: lane=col, pass the
// "N-major" pointer). Per ISA 7.12.2: lanes 0-15 hold k-chunks {0..7,16..23},
// lanes 16-31 hold {8..15,24..31}. Works for global and LDS pointers.
static __device__ __forceinline__ v16h frag_h(const _Float16* p, int stride) {
    int lane = threadIdx.x & 31;
    int row  = lane & 15;
    int k0   = (lane >> 4) << 3;
    const _Float16* r0 = p + (size_t)row * stride + k0;
    union { v16h v; v8h h[2]; } u;
    u.h[0] = *(const v8h*)(r0);
    u.h[1] = *(const v8h*)(r0 + 16);
    return u.v;
}

// Same fragment, sourced from f32 memory with inline convert.
static __device__ __forceinline__ v16h frag_f32(const float* p, int stride) {
    int lane = threadIdx.x & 31;
    int row  = lane & 15;
    int k0   = (lane >> 4) << 3;
    const float* r0 = p + (size_t)row * stride + k0;
    union { v16h v; _Float16 e[16]; } u;
#pragma unroll
    for (int i = 0; i < 8; ++i) {
        u.e[i]     = (_Float16)r0[i];
        u.e[8 + i] = (_Float16)r0[16 + i];
    }
    return u.v;
}

static __device__ __forceinline__ float rmax16(float x) {
#pragma unroll
    for (int m = 1; m < 16; m <<= 1) x = fmaxf(x, __shfl_xor(x, m, 32));
    return x;
}
static __device__ __forceinline__ float rsum16(float x) {
#pragma unroll
    for (int m = 1; m < 16; m <<= 1) x += __shfl_xor(x, m, 32);
    return x;
}

// Async DMA of one 16-byte chunk from global to LDS (GVS mode, ASYNCcnt).
static __device__ __forceinline__ void async_copy16(uint32_t lds_byte_addr,
                                                    uint32_t g_byte_off,
                                                    const void* s_base) {
    asm volatile("global_load_async_to_lds_b128 %0, %1, %2"
                 :: "v"(lds_byte_addr), "v"(g_byte_off), "s"(s_base)
                 : "memory");
}

// ---------------- Kernel 1: GroupNorm statistics -------------------------
__global__ void gn_stats_kernel(const float* __restrict__ x, float* __restrict__ stats) {
    int bg = blockIdx.x;  // b*8 + g
    const float4* p = (const float4*)(x + (size_t)bg * (32 * NN));
    float s = 0.f, q = 0.f;
    for (int i = threadIdx.x; i < (32 * NN) / 4; i += 256) {
        float4 v = p[i];
        s += v.x + v.y + v.z + v.w;
        q += v.x * v.x + v.y * v.y + v.z * v.z + v.w * v.w;
    }
    __shared__ float sh[512];
    sh[threadIdx.x]       = s;
    sh[256 + threadIdx.x] = q;
    __syncthreads();
    for (int st = 128; st > 0; st >>= 1) {
        if ((int)threadIdx.x < st) {
            sh[threadIdx.x]       += sh[threadIdx.x + st];
            sh[256 + threadIdx.x] += sh[256 + threadIdx.x + st];
        }
        __syncthreads();
    }
    if (threadIdx.x == 0) {
        const float cnt = 32.f * NN;
        float mean = sh[0] / cnt;
        float var  = sh[256] / cnt - mean * mean;
        stats[bg * 2]     = mean;
        stats[bg * 2 + 1] = rsqrtf(var + 1e-5f);
    }
}

// ---------------- Kernel 2: fused GroupNorm + QKV GEMM -------------------
// grid (12 o-tiles of 64, 32 p-tiles of 128, b). 256 threads = 8 waves.
// Writes q,k as [bh][n][64] f16 and v as [bh][64][n] f16.
__global__ void qkv_kernel(const float* __restrict__ x,
                           const float* __restrict__ gn_w,
                           const float* __restrict__ gn_b,
                           const float* __restrict__ qkv_w,
                           const float* __restrict__ qkv_b,
                           const float* __restrict__ stats,
                           _Float16* __restrict__ qbuf,
                           _Float16* __restrict__ kbuf,
                           _Float16* __restrict__ vbuf) {
    __shared__ __align__(16) _Float16 w_tile[64 * 32];    // [o_local][k_local]
    __shared__ __align__(16) _Float16 xn_tile[128 * 32];  // [p_local][k_local] (transposed)
    int tid = threadIdx.x;
    int o_base = blockIdx.x * 64;
    int p_base = blockIdx.y * 128;
    int b = blockIdx.z;
    int wv = tid >> 5, lane = tid & 31;
    int o_sub = wv >> 1, psub0 = (wv & 1) * 4;
    v8f acc[4] = {};
    int wrow = tid >> 2, wcol = (tid & 3) * 8;  // W staging: 64 rows x 4 chunks
    int xc = tid >> 3, xp = (tid & 7) * 16;     // x staging: 32 ch x 8 chunks of 16 px

    for (int kk = 0; kk < CC; kk += 32) {
        {   // stage W (f32 -> f16)
            const float* src = qkv_w + (size_t)(o_base + wrow) * CC + kk + wcol;
#pragma unroll
            for (int i = 0; i < 8; ++i)
                w_tile[wrow * 32 + wcol + i] = (_Float16)src[i];
        }
        {   // stage normalized x, transposed to [p][c]
            int c = kk + xc;
            int g = c >> 5;
            float mean = stats[(b * NGROUP + g) * 2];
            float rstd = stats[(b * NGROUP + g) * 2 + 1];
            float ga = gn_w[c] * rstd;
            float be = gn_b[c] - mean * ga;
            const float* src = x + ((size_t)b * CC + c) * NN + p_base + xp;
#pragma unroll
            for (int i = 0; i < 16; ++i)
                xn_tile[(xp + i) * 32 + xc] = (_Float16)(src[i] * ga + be);
        }
        __syncthreads();
        v16h a = frag_h(w_tile + o_sub * 16 * 32, 32);
#pragma unroll
        for (int t = 0; t < 4; ++t) {
            v16h bb = frag_h(xn_tile + (psub0 + t) * 16 * 32, 32);
            acc[t] = wmma16(a, bb, acc[t]);
        }
        __syncthreads();
    }

    // epilogue: bias, convert, scatter into q/k/v layouts
    int hi = lane >> 4, col = lane & 15;
    int o0 = o_base + o_sub * 16 + hi * 8;      // first of 8 consecutive out channels
    int which = o0 >> 8;                        // 0=q 1=k 2=v
    int oc = o0 & 255;
    int head = oc >> 6, d0 = oc & 63;
    int bh = b * NHEADS + head;
#pragma unroll
    for (int t = 0; t < 4; ++t) {
        int p = p_base + (psub0 + t) * 16 + col;
        v8h e;
#pragma unroll
        for (int r = 0; r < 8; ++r)
            e[r] = (_Float16)(acc[t][r] + qkv_b[o0 + r]);
        if (which == 0) {
            *(v8h*)(qbuf + ((size_t)bh * NN + p) * HD + d0) = e;
        } else if (which == 1) {
            *(v8h*)(kbuf + ((size_t)bh * NN + p) * HD + d0) = e;
        } else {
#pragma unroll
            for (int r = 0; r < 8; ++r)
                vbuf[((size_t)bh * HD + d0 + r) * NN + p] = e[r];
        }
    }
}

// ---------------- Kernel 3: flash attention ------------------------------
// grid (32 q-tiles of 128, 16 bh). 8 waves; each wave owns 16 query rows.
// K/V tiles (32 keys) are staged into LDS once per block via async DMA,
// double-buffered so the DMA for tile i+1 overlaps compute on tile i.
__global__ void flash_kernel(const _Float16* __restrict__ qbuf,
                             const _Float16* __restrict__ kbuf,
                             const _Float16* __restrict__ vbuf,
                             _Float16* __restrict__ aobuf) {
    // per-buffer: K tile [32 keys][64] (4KB) then V tile [64][32 keys] (4KB)
    __shared__ __align__(16) _Float16 lds_kv[2][32 * 64 + 64 * 32];
    __shared__ __align__(16) _Float16 lds_p[8 * 16 * 32];  // per-wave P tile
    __shared__ __align__(16) _Float16 lds_o[8 * 16 * 64];  // per-wave O tile
    int tid = threadIdx.x;
    int wv = tid >> 5, lane = tid & 31;
    int hi = lane >> 4, col = lane & 15;
    int bh = blockIdx.y;
    int q0 = blockIdx.x * 128 + wv * 16;

    const _Float16* qp    = qbuf + ((size_t)bh * NN + q0) * HD;
    const _Float16* kbase = kbuf + (size_t)bh * NN * HD;   // [n][64]
    const _Float16* vbase = vbuf + (size_t)bh * HD * NN;   // [64][n]
    _Float16* ldsp = lds_p + wv * (16 * 32);
    _Float16* ldso = lds_o + wv * (16 * 64);

    // async staging: 256 threads x 16B each for K (32x8 chunks) and V (64x4)
    int krow = tid >> 3, kch = tid & 7;
    int vrow = tid >> 2, vch = tid & 3;
    uint32_t k_lds[2], v_lds[2];
#pragma unroll
    for (int s = 0; s < 2; ++s) {
        k_lds[s] = (uint32_t)(uintptr_t)(&lds_kv[s][0] + krow * 64 + kch * 8);
        v_lds[s] = (uint32_t)(uintptr_t)(&lds_kv[s][32 * 64] + vrow * 32 + vch * 8);
    }
    const uint32_t k_goff = (uint32_t)((krow * HD + kch * 8) * 2);
    const uint32_t v_goff = (uint32_t)((vrow * NN + vch * 8) * 2);

    v16h aq0 = frag_h(qp, HD);        // d = 0..31
    v16h aq1 = frag_h(qp + 32, HD);   // d = 32..63

    float m[8], l[8];
#pragma unroll
    for (int r = 0; r < 8; ++r) { m[r] = -1e30f; l[r] = 0.f; }
    v8f acc[4] = {};
    const float SC = 0.125f * 1.44269504088896f;  // head_dim^-0.5 * log2(e)

    // prologue: DMA tile 0 into buffer 0
    async_copy16(k_lds[0], k_goff, kbase);
    async_copy16(v_lds[0], v_goff, vbase);

    const int ITERS = NN / 32;
    for (int it = 0; it < ITERS; ++it) {
        int s = it & 1;
        asm volatile("s_wait_asynccnt 0x0" ::: "memory");  // own DMA landed
        __syncthreads();                                   // everyone's DMA landed
        if (it + 1 < ITERS) {                              // overlap next DMA
            int j1 = (it + 1) * 32;
            async_copy16(k_lds[s ^ 1], k_goff + (uint32_t)(j1 * HD * 2), kbase);
            async_copy16(v_lds[s ^ 1], v_goff + (uint32_t)(j1 * 2), vbase);
        }
        const _Float16* kt = &lds_kv[s][0];        // [32 keys][64]
        const _Float16* vt = &lds_kv[s][32 * 64];  // [64][32 keys]

        v16h b00 = frag_h(kt, 64);
        v16h b01 = frag_h(kt + 32, 64);
        v16h b10 = frag_h(kt + 16 * 64, 64);
        v16h b11 = frag_h(kt + 16 * 64 + 32, 64);
        v8f s0 = {}, s1 = {};
        s0 = wmma16(aq0, b00, s0);
        s0 = wmma16(aq1, b01, s0);
        s1 = wmma16(aq0, b10, s1);
        s1 = wmma16(aq1, b11, s1);

#pragma unroll
        for (int r = 0; r < 8; ++r) {
            float a0 = s0[r] * SC, a1 = s1[r] * SC;     // log2 domain
            float mx = rmax16(fmaxf(a0, a1));
            float mn = fmaxf(m[r], mx);
            float p0v = exp2f(a0 - mn), p1v = exp2f(a1 - mn);
            float corr = exp2f(m[r] - mn);
            m[r] = mn;
            l[r] = l[r] * corr + rsum16(p0v + p1v);
            acc[0][r] *= corr; acc[1][r] *= corr;
            acc[2][r] *= corr; acc[3][r] *= corr;
            int row = r + hi * 8;
            ldsp[row * 32 + col]      = (_Float16)p0v;
            ldsp[row * 32 + col + 16] = (_Float16)p1v;
        }
        asm volatile("s_wait_dscnt 0" ::: "memory");
        v16h ap = frag_h(ldsp, 32);   // P as A operand (16x32)
#pragma unroll
        for (int t = 0; t < 4; ++t) {
            v16h bv = frag_h(vt + t * 16 * 32, 32);
            acc[t] = wmma16(ap, bv, acc[t]);
        }
    }

    // normalize + write O via LDS into [b][p][c] f16 layout
#pragma unroll
    for (int r = 0; r < 8; ++r) {
        float inv = 1.0f / l[r];
        int row = r + hi * 8;
#pragma unroll
        for (int t = 0; t < 4; ++t)
            ldso[row * 64 + t * 16 + col] = (_Float16)(acc[t][r] * inv);
    }
    asm volatile("s_wait_dscnt 0" ::: "memory");
    int row = col;                    // query row within wave tile
    int p = q0 + row;
    int b_ = bh >> 2, h_ = bh & 3;
    _Float16* dst = aobuf + ((size_t)b_ * NN + p) * CC + h_ * HD + hi * 32;
    const _Float16* src = ldso + row * 64 + hi * 32;
#pragma unroll
    for (int i = 0; i < 4; ++i)
        *(v8h*)(dst + i * 8) = *(const v8h*)(src + i * 8);
}

// ---------------- Kernel 4: proj GEMM + bias + residual ------------------
// grid (4 o-tiles of 64, 32 p-tiles of 128, b). B operand read straight from
// the [p][c] f16 attention output; A converted f32->f16 in registers.
__global__ void proj_kernel(const _Float16* __restrict__ aobuf,
                            const float* __restrict__ proj_w,
                            const float* __restrict__ proj_b,
                            const float* __restrict__ x,
                            float* __restrict__ out) {
    int tid = threadIdx.x;
    int wv = tid >> 5, lane = tid & 31;
    int o_sub = wv >> 1, psub0 = (wv & 1) * 4;
    int o_base = blockIdx.x * 64 + o_sub * 16;
    int p_base = blockIdx.y * 128;
    int b = blockIdx.z;
    v8f acc[4] = {};
    for (int kk = 0; kk < CC; kk += 32) {
        v16h a = frag_f32(proj_w + (size_t)o_base * CC + kk, CC);
#pragma unroll
        for (int t = 0; t < 4; ++t) {
            v16h bb = frag_h(aobuf + ((size_t)b * NN + p_base + (psub0 + t) * 16) * CC + kk, CC);
            acc[t] = wmma16(a, bb, acc[t]);
        }
    }
    int hi = lane >> 4, col = lane & 15;
    int o0 = o_base + hi * 8;
#pragma unroll
    for (int t = 0; t < 4; ++t) {
        int p = p_base + (psub0 + t) * 16 + col;
#pragma unroll
        for (int r = 0; r < 8; ++r) {
            int o = o0 + r;
            size_t idx = ((size_t)b * CC + o) * NN + p;
            out[idx] = acc[t][r] + proj_b[o] + x[idx];
        }
    }
}

extern "C" void kernel_launch(void* const* d_in, const int* in_sizes, int n_in,
                              void* d_out, int out_size, void* d_ws, size_t ws_size,
                              hipStream_t stream) {
    const float* x      = (const float*)d_in[0];
    const float* gn_w   = (const float*)d_in[1];
    const float* gn_b   = (const float*)d_in[2];
    const float* qkv_w  = (const float*)d_in[3];
    const float* qkv_b  = (const float*)d_in[4];
    const float* proj_w = (const float*)d_in[5];
    const float* proj_b = (const float*)d_in[6];
    float* out = (float*)d_out;

    // workspace layout
    float* stats = (float*)d_ws;                                   // 64 floats
    const size_t QKV_ELEMS = (size_t)BB * NHEADS * NN * HD;        // 4 Mi f16
    _Float16* qbuf  = (_Float16*)((char*)d_ws + 1024);
    _Float16* kbuf  = qbuf + QKV_ELEMS;
    _Float16* vbuf  = kbuf + QKV_ELEMS;
    _Float16* aobuf = vbuf + QKV_ELEMS;                            // [b][n][c]

    gn_stats_kernel<<<BB * NGROUP, 256, 0, stream>>>(x, stats);
    qkv_kernel<<<dim3(12, 32, BB), 256, 0, stream>>>(x, gn_w, gn_b, qkv_w, qkv_b,
                                                     stats, qbuf, kbuf, vbuf);
    flash_kernel<<<dim3(32, BB * NHEADS), 256, 0, stream>>>(qbuf, kbuf, vbuf, aobuf);
    proj_kernel<<<dim3(4, 32, BB), 256, 0, stream>>>(aobuf, proj_w, proj_b, x, out);
}
